// MoELayerStacks_29600914604664
// MI455X (gfx1250) — matrix operand order repaired and checked by reading, not compiled, not run on hardware
//
#include <hip/hip_runtime.h>
#include <hip/hip_bf16.h>

// ---- problem constants --------------------------------------------------
#define E_    8
#define L1_   3072
#define L2_   15
#define L3_   32
#define B_    16384
#define N1    128            // E * (L2+1) total layer-1 outputs
#define TILE_M 128
#define TILE_K 32
#define KSTEPS (L1_ / TILE_K) // 96
#define LDA   36             // LDS stride (halves) for A tile: 72B rows, 8B-aligned stores
#define LDB   40             // LDS stride (halves) for B tile: 80B rows, 16B-aligned async stores
#define LDC   132            // LDS stride (floats) for the l1c tile, padded

typedef __attribute__((ext_vector_type(16))) __bf16 v16bf;
typedef __attribute__((ext_vector_type(8)))  float  v8f;

union Frag16 { v16bf v; unsigned int u[8]; };

__device__ __forceinline__ float clip01(float v) {
  return fminf(fmaxf(v, 0.0f), 1.0f);
}

// gfx1250 async copy: global memory -> LDS, tracked by ASYNCcnt.
// vdst = LDS byte offset (low 32 bits of generic shared pointer), vaddr = 64b global.
__device__ __forceinline__ void async_g2l_b128(unsigned lds_off, const void* gptr) {
  asm volatile("global_load_async_to_lds_b128 %0, %1, off"
               :: "v"(lds_off), "v"(gptr) : "memory");
}
__device__ __forceinline__ void wait_asynccnt0() {
  asm volatile("s_wait_asynccnt 0" ::: "memory");
}

// ---- pre-pass: convert w1 (128 x 3072 f32, row = output column) to bf16 --
__global__ __launch_bounds__(256)
void convert_w1_kernel(const float* __restrict__ w1,
                       unsigned short* __restrict__ dst)
{
  size_t base = ((size_t)blockIdx.x * 256 + threadIdx.x) * 4;   // 98304 threads * 4
  float4 f = *(const float4*)(w1 + base);
  union { __bf16 h[4]; uint2 u; } p;
  p.h[0] = (__bf16)f.x; p.h[1] = (__bf16)f.y;
  p.h[2] = (__bf16)f.z; p.h[3] = (__bf16)f.w;
  *(uint2*)(dst + base) = p.u;
}

// ---- fused kernel: layer1 WMMA GEMM + expert MLPs + router/losses -------
__global__ __launch_bounds__(256)
void moe_fused_kernel(const float* __restrict__ x,
                      const float* __restrict__ psqt,
                      const float* __restrict__ tgt,
                      const float* __restrict__ rW,
                      const float* __restrict__ rB,
                      const unsigned short* __restrict__ w1bf,
                      const float* __restrict__ b1,
                      const float* __restrict__ w2,
                      const float* __restrict__ b2,
                      const float* __restrict__ w3,
                      const float* __restrict__ b3,
                      float* __restrict__ out,
                      float* __restrict__ ws)
{
  // dynamic LDS: [0,9216) lA | [9216,19456) lB       during GEMM
  //              [0,67584) sC (128x132 f32)          after GEMM (aliases lA/lB)
  //              [67584,71680) sOut (128x8 f32)
  extern __shared__ __align__(16) char smem[];
  unsigned short* lA  = (unsigned short*)smem;
  unsigned short* lB  = (unsigned short*)(smem + 9216);
  float*          sC  = (float*)smem;
  float*          sOut= (float*)(smem + 67584);

  __shared__ float sW2[E_ * L3_ * 2 * L2_];  // 7680
  __shared__ float sB2[E_ * L3_];
  __shared__ float sW3[E_ * L3_];
  __shared__ float sB3[E_];
  __shared__ float sRW[E_ * 64];
  __shared__ float sRB[E_];
  __shared__ float sB1[N1];
  __shared__ int   sCnt[E_];
  __shared__ float sZ[128];
  __shared__ float sP[128];

  const int tid  = threadIdx.x;
  const int lane = tid & 31;
  const int wave = tid >> 5;
  const int wm   = (wave & 3) * 32;   // wave M offset (4 waves along M)
  const int wn   = (wave >> 2) * 64;  // wave N offset (2 waves along N)
  const int hi   = lane >> 4;
  const int lo   = lane & 15;

  // ---- one-time staging of the small weights ---------------------------
  for (int i = tid; i < E_ * L3_ * 2 * L2_; i += 256) sW2[i] = w2[i];
  for (int i = tid; i < E_ * L3_; i += 256) { sB2[i] = b2[i]; sW3[i] = w3[i]; }
  for (int i = tid; i < E_ * 64; i += 256) sRW[i] = rW[i];
  for (int i = tid; i < N1; i += 256) sB1[i] = b1[i];
  if (tid < E_) { sB3[tid] = b3[tid]; sRB[tid] = rB[tid]; sCnt[tid] = 0; }

  const float* xblk = x + (size_t)blockIdx.x * TILE_M * L1_;

  // ---- GEMM: C[128x128] = x_tile[128x3072] * w1^T[3072x128] (bf16 WMMA) -
  v8f acc[2][4];
  #pragma unroll
  for (int mi = 0; mi < 2; ++mi)
    #pragma unroll
    for (int ni = 0; ni < 4; ++ni)
      #pragma unroll
      for (int r = 0; r < 8; ++r) acc[mi][ni][r] = 0.0f;

  for (int ks = 0; ks < KSTEPS; ++ks) {
    const int k0 = ks * TILE_K;
    __syncthreads();                        // prior compute done reading LDS

    // B tile (already bf16): async copy 128x32 halves, 2 x b128 per thread
    #pragma unroll
    for (int i = 0; i < 2; ++i) {
      int lin = tid + 256 * i;              // 0..511
      int n  = lin >> 2;                    // 0..127 output column
      int cg = (lin & 3) * 8;               // half offset: 0,8,16,24
      async_g2l_b128((unsigned)(size_t)(void*)(lB + n * LDB + cg),
                     w1bf + (size_t)n * L1_ + k0 + cg);
    }

    // A tile: vectorized f32 load + packed bf16 convert + b64 LDS store
    #pragma unroll
    for (int i = 0; i < 4; ++i) {
      int lin = tid + 256 * i;              // 0..1023
      int r  = lin >> 3;                    // row 0..127
      int cq = (lin & 7) * 4;               // col 0,4,...,28
      const float4 f = *(const float4*)(xblk + (size_t)r * L1_ + k0 + cq);
      union { __bf16 h[4]; uint2 u; } p;
      p.h[0] = (__bf16)f.x; p.h[1] = (__bf16)f.y;
      p.h[2] = (__bf16)f.z; p.h[3] = (__bf16)f.w;
      *(uint2*)(lA + r * LDA + cq) = p.u;
    }

    // gfx1250 prefetch of next x K-tile (one cacheline per row)
    if (ks + 1 < KSTEPS && tid < 128)
      __builtin_prefetch(&xblk[(size_t)tid * L1_ + k0 + TILE_K], 0, 1);

    wait_asynccnt0();                       // our async LDS writes landed
    __syncthreads();

    // fragment loads per ISA 16-bit A/B layouts (dword = 2 consecutive K)
    Frag16 af[2], bfc[4];
    #pragma unroll
    for (int mi = 0; mi < 2; ++mi) {
      const unsigned short* rp = lA + (wm + mi * 16 + lo) * LDA;
      #pragma unroll
      for (int v = 0; v < 8; ++v) {
        int k = ((v >= 4) ? 16 : 0) + ((v & 3) << 1) + hi * 8;
        af[mi].u[v] = *(const unsigned int*)(rp + k);
      }
    }
    #pragma unroll
    for (int ni = 0; ni < 4; ++ni) {
      const unsigned short* cp = lB + (wn + ni * 16 + lo) * LDB;
      #pragma unroll
      for (int v = 0; v < 8; ++v) {
        int k = ((v >= 4) ? 16 : 0) + ((v & 3) << 1) + hi * 8;
        bfc[ni].u[v] = *(const unsigned int*)(cp + k);
      }
    }
    #pragma unroll
    for (int mi = 0; mi < 2; ++mi)
      #pragma unroll
      for (int ni = 0; ni < 4; ++ni)
        acc[mi][ni] = __builtin_amdgcn_wmma_f32_16x16x32_bf16(
            false, af[mi].v, false, bfc[ni].v, (short)0, acc[mi][ni],
            false, false);
  }
  __syncthreads();   // done with lA/lB; sC aliases their storage

  // ---- epilogue: l1c tile (+bias) into LDS ------------------------------
  #pragma unroll
  for (int mi = 0; mi < 2; ++mi)
    #pragma unroll
    for (int ni = 0; ni < 4; ++ni)
      #pragma unroll
      for (int r = 0; r < 8; ++r) {
        int M = wm + mi * 16 + hi * 8 + r;
        int N = wn + ni * 16 + lo;
        sC[M * LDC + N] = acc[mi][ni][r] + sB1[N];
      }
  __syncthreads();

  // ---- per-(row, expert) MLP: h -> l2 -> all_out ------------------------
  for (int it = 0; it < 4; ++it) {
    int task = tid + 256 * it;              // 0..1023 = 128 rows * 8 experts
    int row = task >> 3, e = task & 7;
    const float* c = sC + row * LDC + e * 16;
    float h[30];
    #pragma unroll
    for (int i = 0; i < 15; ++i) {
      float a = c[i];
      h[i]      = clip01(a * a * (255.0f / 256.0f));
      h[15 + i] = clip01(a);
    }
    float askip = c[15];
    const float* w2e = sW2 + e * L3_ * 30;
    const float* w3e = sW3 + e * L3_;
    float dot3 = 0.0f;
    #pragma unroll 4
    for (int o = 0; o < 32; ++o) {
      float s = sB2[e * 32 + o];
      const float* w = w2e + o * 30;
      #pragma unroll
      for (int i = 0; i < 30; ++i) s += h[i] * w[i];
      dot3 += clip01(s) * w3e[o];
    }
    sOut[row * 8 + e] = dot3 + sB3[e] + askip;
  }
  __syncthreads();

  // ---- per-row router, argmax pick, loss partials -----------------------
  if (tid < 128) {
    int gb = blockIdx.x * 128 + tid;
    const float* xr = x + (size_t)gb * L1_;
    float logit[8];
    #pragma unroll
    for (int e = 0; e < 8; ++e) logit[e] = sRB[e];
    for (int j = 0; j < 32; ++j) {
      float v0 = xr[j], v1 = xr[1536 + j];
      #pragma unroll
      for (int e = 0; e < 8; ++e)
        logit[e] += v0 * sRW[e * 64 + j] + v1 * sRW[e * 64 + 32 + j];
    }
    float mx = logit[0]; int am = 0;
    #pragma unroll
    for (int e = 1; e < 8; ++e) if (logit[e] > mx) { mx = logit[e]; am = e; }
    float se = 0.0f;
    #pragma unroll
    for (int e = 0; e < 8; ++e) se += __expf(logit[e] - mx);
    float lse = mx + __logf(se);

    float ps = psqt[gb], tg = tgt[gb];
    float err[8]; float emin = 3.4e38f;
    #pragma unroll
    for (int e = 0; e < 8; ++e) {
      float d = (sOut[tid * 8 + e] + ps) * 600.0f - tg;
      err[e] = d * d;
      emin = fminf(emin, err[e]);
    }
    float pu[8]; float pden = 0.0f;
    #pragma unroll
    for (int e = 0; e < 8; ++e) { pu[e] = __expf(-(err[e] - emin)); pden += pu[e]; }
    float probe = 0.0f;
    #pragma unroll
    for (int e = 0; e < 8; ++e) {
      float pt  = pu[e] / pden;
      float lpt = __logf(fmaxf(pt, 1e-30f));
      probe += pt * (lpt - (logit[e] - lse));
    }
    out[gb] = sOut[tid * 8 + am];           // forward gating factor == 1.0
    sZ[tid] = lse * lse;
    sP[tid] = probe;
    atomicAdd(&sCnt[am], 1);                // integer: order-independent
  }
  __syncthreads();
  // deterministic fixed-pairing tree reduction over 128 rows
  for (int s = 64; s > 0; s >>= 1) {
    if (tid < s) { sZ[tid] += sZ[tid + s]; sP[tid] += sP[tid + s]; }
    __syncthreads();
  }
  if (tid == 0) {
    float* p = ws + blockIdx.x * 10;
    #pragma unroll
    for (int e = 0; e < 8; ++e) p[e] = (float)sCnt[e];
    p[8] = sZ[0];
    p[9] = sP[0];
  }
}

// ---- tiny finalize: fold 128 block partials into the scalar loss --------
__global__ void moe_finalize_kernel(const float* __restrict__ ws,
                                    float* __restrict__ out)
{
  if (threadIdx.x != 0 || blockIdx.x != 0) return;
  float cnt[8] = {0,0,0,0,0,0,0,0};
  float z = 0.0f, p = 0.0f;
  for (int b = 0; b < 128; ++b) {
    const float* q = ws + b * 10;
    #pragma unroll
    for (int e = 0; e < 8; ++e) cnt[e] += q[e];
    z += q[8];
    p += q[9];
  }
  float aux = 0.0f;
  #pragma unroll
  for (int e = 0; e < 8; ++e) {
    float f  = cnt[e] * (1.0f / 16384.0f);   // load == frac in forward
    float fl = fmaxf(0.05f - f, 0.0f);
    float cp = fmaxf(f - 0.5f, 0.0f);
    aux += fl * fl + cp * cp;
  }
  aux *= (1.0f / 8.0f);                      // floor_loss + cap_loss (means)
  float zl = z * (1.0f / 16384.0f);
  float pl = p * (1.0f / 16384.0f);
  // probe_weight = 0.01 * max(0.05-0,0)/0.05 = 0.01 ; TEACHER_ALPHA = 0
  out[16384] = 0.01f * aux + 0.001f * zl + 0.01f * pl;
}

extern "C" void kernel_launch(void* const* d_in, const int* in_sizes, int n_in,
                              void* d_out, int out_size, void* d_ws, size_t ws_size,
                              hipStream_t stream)
{
  (void)in_sizes; (void)n_in; (void)out_size; (void)ws_size;
  const float* x    = (const float*)d_in[0];
  // d_in[1] = ls_indices (int32) unused: TEACHER_ALPHA == 0
  const float* psqt = (const float*)d_in[2];
  const float* tgt  = (const float*)d_in[3];
  const float* rW   = (const float*)d_in[4];
  const float* rB   = (const float*)d_in[5];
  const float* w1   = (const float*)d_in[6];
  const float* b1   = (const float*)d_in[7];
  const float* w2   = (const float*)d_in[8];
  const float* b2   = (const float*)d_in[9];
  const float* w3   = (const float*)d_in[10];
  const float* b3   = (const float*)d_in[11];
  float* out = (float*)d_out;

  // d_ws layout: [0, 786432) bf16 copy of w1 ; [786432, 791552) block partials
  unsigned short* w1bf = (unsigned short*)d_ws;
  float* partials = (float*)((char*)d_ws + 786432);

  convert_w1_kernel<<<384, 256, 0, stream>>>(w1, w1bf);

  const size_t dyn_lds = 67584 + 4096;   // sC (128x132 f32) + sOut (128x8 f32)
  moe_fused_kernel<<<B_ / TILE_M, 256, dyn_lds, stream>>>(
      x, psqt, tgt, rW, rB, w1bf, b1, w2, b2, w3, b3, out, partials);
  moe_finalize_kernel<<<1, 32, 0, stream>>>(partials, out);
}
